// SA_Block_CrossPoolKV_38130719654099
// MI455X (gfx1250) — compile-verified, matrix operand order checked by reading, not run on hardware
//
#include <hip/hip_runtime.h>

// ---------------------------------------------------------------------------
// CDNA5 (gfx1250) fused SA block with cross-pooled KV.
// f16 WMMA (v_wmma_f32_16x16x32_f16) with fp32 accumulation, flash-attention
// online softmax held in registers (shfl_xor butterflies over the D-layout
// column axis). All WMMA operands are K-contiguous: every fragment is two
// 16-byte b128 loads. Only the P (D->A layout) transpose touches LDS.
// ---------------------------------------------------------------------------

typedef _Float16 f16;
typedef __attribute__((ext_vector_type(16))) _Float16 v16h;
typedef __attribute__((ext_vector_type(8)))  float    v8f;

union F16Frag { v16h v; _Float16 e[16]; float4 q4[2]; };
union F16x8   { float4 q; _Float16 e[8]; };
union F32Acc  { v8f  v; float    e[8]; };

__device__ __forceinline__ v8f wmma_f16(v16h a, v16h b, v8f c) {
  return __builtin_amdgcn_wmma_f32_16x16x32_f16(false, a, false, b, (short)0, c,
                                                false, false);
}

// Fragment load when the K axis is contiguous in memory: two 16B loads.
// Halves 0..7 <- rowptr[lh*8 .. lh*8+7], halves 8..15 <- rowptr[16+lh*8 ..].
__device__ __forceinline__ v16h load_frag_kcontig(const f16* rowptr, int lh) {
  F16Frag f;
  const f16* p = rowptr + (lh << 3);
  f.q4[0] = *(const float4*)(p);
  f.q4[1] = *(const float4*)(p + 16);
  return f.v;
}

// Butterfly reductions across the 16 lanes of a lane-half (masks 1,2,4,8 keep
// the exchange within each half; result uniform across the half).
__device__ __forceinline__ float red16_max(float v) {
  v = fmaxf(v, __shfl_xor(v, 1));
  v = fmaxf(v, __shfl_xor(v, 2));
  v = fmaxf(v, __shfl_xor(v, 4));
  v = fmaxf(v, __shfl_xor(v, 8));
  return v;
}
__device__ __forceinline__ float red16_sum(float v) {
  v += __shfl_xor(v, 1);
  v += __shfl_xor(v, 2);
  v += __shfl_xor(v, 4);
  v += __shfl_xor(v, 8);
  return v;
}

// ---------------------------------------------------------------------------
// Common flash-attention inner loop (one wave, one 16-query tile).
// KbT: (NKEYS, 64) K-transposed; Vb: (64, NKEYS). sPw: this wave's 16x32 LDS
// staging for the P fragment transpose. oAcc: 16x64 output tile (4 D tiles).
template <int NKEYS>
__device__ __forceinline__ void flash_loop(const f16* __restrict__ KbT,
                                           const f16* __restrict__ Vb,
                                           v16h aq0, v16h aq1,
                                           f16 (*__restrict__ sPw)[32],
                                           F32Acc* __restrict__ oAcc,
                                           float* __restrict__ lrun,
                                           int lr, int lh) {
  float mrun[8];
#pragma unroll
  for (int r = 0; r < 8; ++r) { mrun[r] = -1e30f; lrun[r] = 0.0f; }
  for (int m0 = 0; m0 < NKEYS; m0 += 32) {
    if (m0 + 32 < NKEYS)                     // prefetch next K tile (one row/lane)
      __builtin_prefetch(KbT + (size_t)(m0 + 32 + lr + (lh << 4)) * 64, 0, 1);
    // S chunk 16x32 as two 16x16 D tiles, K=c accumulated over two WMMAs each.
    const int mcol0 = m0 + lr, mcol1 = m0 + 16 + lr;
    const v16h bk00 = load_frag_kcontig(KbT + (size_t)mcol0 * 64, lh);
    const v16h bk01 = load_frag_kcontig(KbT + (size_t)mcol0 * 64 + 32, lh);
    const v16h bk10 = load_frag_kcontig(KbT + (size_t)mcol1 * 64, lh);
    const v16h bk11 = load_frag_kcontig(KbT + (size_t)mcol1 * 64 + 32, lh);
    F32Acc s0 = {}, s1 = {};
    s0.v = wmma_f16(aq0, bk00, s0.v);
    s0.v = wmma_f16(aq1, bk01, s0.v);
    s1.v = wmma_f16(aq0, bk10, s1.v);
    s1.v = wmma_f16(aq1, bk11, s1.v);
    // Online softmax fully in registers: row (r + lh*8) stats are uniform
    // across each 16-lane half after the butterflies.
    float alpha[8];
#pragma unroll
    for (int r = 0; r < 8; ++r) {
      const float mc = red16_max(fmaxf(s0.e[r], s1.e[r]));
      const float mnew = fmaxf(mrun[r], mc);
      alpha[r] = __expf(mrun[r] - mnew);
      mrun[r] = mnew;
      const float p0 = __expf(s0.e[r] - mnew);
      const float p1 = __expf(s1.e[r] - mnew);
      sPw[r + (lh << 3)][lr]      = (f16)p0;   // D->A layout transpose via LDS
      sPw[r + (lh << 3)][16 + lr] = (f16)p1;
      lrun[r] = lrun[r] * alpha[r] + red16_sum(p0 + p1);
    }
    __builtin_amdgcn_wave_barrier();           // keep LDS stores before reload
    const v16h ap = load_frag_kcontig(&sPw[lr][0], lh);
#pragma unroll
    for (int ct = 0; ct < 4; ++ct) {           // O += P @ V^T
      const v16h bv =
          load_frag_kcontig(Vb + (size_t)((ct << 4) + lr) * NKEYS + m0, lh);
#pragma unroll
      for (int r = 0; r < 8; ++r) oAcc[ct].e[r] *= alpha[r];
      oAcc[ct].v = wmma_f16(ap, bv, oAcc[ct].v);
    }
    __builtin_amdgcn_wave_barrier();           // sPw reused next iteration
  }
}

// ---------------------------------------------------------------------------
__global__ void k_pack_f16(const float* __restrict__ src, f16* __restrict__ dst,
                           int n) {
  int i = blockIdx.x * 256 + threadIdx.x;
  if (i < n) dst[i] = (f16)src[i];
}

// x (B,64,4096) f32 -> xT (B,4096,64) f16  (write-coalesced transpose+convert)
__global__ void k_pack_xT(const float* __restrict__ x, f16* __restrict__ xT) {
  int i = blockIdx.x * 256 + threadIdx.x;           // (b, n, c), c fastest
  const int c = i & 63, n = (i >> 6) & 4095, b = i >> 18;
  xT[i] = (f16)x[((size_t)(b * 64 + c) << 12) + n];
}

// ---------------------------------------------------------------------------
// Y = W @ X + bias per batch. Xt: (B,N,64) f16 (K=c contiguous). W: 64x64.
// transposeOut=1 -> Y (B,N,64); transposeOut=0 -> Y (B,64,N).
// Block 256 = 8 waves; each wave owns one 16-column n-tile, all 64 outputs.
__global__ void k_conv1x1(const f16* __restrict__ Xt, const f16* __restrict__ W,
                          const float* __restrict__ bias, f16* __restrict__ Y,
                          int N, int transposeOut) {
  const int lane = threadIdx.x & 31, lr = lane & 15, lh = lane >> 4;
  const int wave = threadIdx.x >> 5;
  const int b = blockIdx.y;
  const int n0 = ((blockIdx.x << 3) + wave) << 4;
  const f16* Xb = Xt + ((size_t)b * N + n0) * 64;
  const v16h b0 = load_frag_kcontig(Xb + lr * 64, lh);
  const v16h b1 = load_frag_kcontig(Xb + lr * 64 + 32, lh);
#pragma unroll
  for (int ot = 0; ot < 4; ++ot) {
    const int obase = ot << 4;
    const v16h a0 = load_frag_kcontig(W + (obase + lr) * 64, lh);
    const v16h a1 = load_frag_kcontig(W + (obase + lr) * 64 + 32, lh);
    F32Acc acc = {};
    acc.v = wmma_f16(a0, b0, acc.v);
    acc.v = wmma_f16(a1, b1, acc.v);
    if (transposeOut) {
      // element (M=o, N=n) -> Y[(b*N + n)*64 + o]; 8 contiguous f16 per lane.
      F16x8 st;
#pragma unroll
      for (int r = 0; r < 8; ++r)
        st.e[r] = (f16)(acc.e[r] + bias[obase + (lh << 3) + r]);
      *(float4*)(Y + ((size_t)b * N + n0 + lr) * 64 + obase + (lh << 3)) = st.q;
    } else {
#pragma unroll
      for (int r = 0; r < 8; ++r) {
        const int o = obase + (lh << 3) + r;
        Y[((size_t)(b * 64 + o)) * N + n0 + lr] = (f16)(acc.e[r] + bias[o]);
      }
    }
  }
}

// ---------------------------------------------------------------------------
// 2x2 avg-pool of qT (B,4096,64) -> qp (B,1024,64). Fully coalesced (c fastest).
__global__ void k_pool_q(const f16* __restrict__ qT, f16* __restrict__ qp) {
  int i = blockIdx.x * 256 + threadIdx.x;           // (b, p, c), c fastest
  if (i >= 8 * 1024 * 64) return;
  const int c = i & 63, p = (i >> 6) & 1023, b = i >> 16;
  const int pi = p >> 5, pj = p & 31;
  const int n00 = (pi << 7) + (pj << 1);
  const f16* qb = qT + (((size_t)(b << 12) + n00) << 6) + c;
  const float v = ((float)qb[0] + (float)qb[64] +
                   (float)qb[64 * 64] + (float)qb[64 * 65]) * 0.25f;
  qp[i] = (f16)v;
}

// ---------------------------------------------------------------------------
// Stage 1 flash attention. qp (B,1024,64), KT (B,4096,64), V (B,64,4096).
// out1T[b,n4,c] = sum_m V[b,c,m] * softmax_m(qp[b,n4,:] . KT[b,m,:]).
__global__ void k_attn_stage1(const f16* __restrict__ qp,
                              const f16* __restrict__ kT,
                              const f16* __restrict__ vf,
                              f16* __restrict__ out1T) {
  __shared__ __attribute__((aligned(16))) f16 sP[8][16][32];
  const int wave = threadIdx.x >> 5, lane = threadIdx.x & 31;
  const int lr = lane & 15, lh = lane >> 4;
  const int b  = blockIdx.y;
  const int qt = (blockIdx.x << 3) + wave;                 // 0..63
  const f16* Qb  = qp + (size_t)(b * 1024 + qt * 16) * 64;
  const f16* KbT = kT + (size_t)b * 4096 * 64;             // [m][c]
  const f16* Vb  = vf + (size_t)b * 64 * 4096;             // [c][m]
  const v16h aq0 = load_frag_kcontig(Qb + lr * 64, lh);
  const v16h aq1 = load_frag_kcontig(Qb + lr * 64 + 32, lh);
  F32Acc oAcc[4] = {};
  float lrun[8];
  flash_loop<4096>(KbT, Vb, aq0, aq1, sP[wave], oAcc, lrun, lr, lh);
  // Store transposed (N4, C) so the stage-2 convs consume it K-contiguously.
  f16* Ob = out1T + (size_t)b * 1024 * 64;
#pragma unroll
  for (int ct = 0; ct < 4; ++ct) {
    const int c = (ct << 4) + lr;
#pragma unroll
    for (int r = 0; r < 8; ++r) {
      const int n = (qt << 4) + (lh << 3) + r;
      Ob[(size_t)n * 64 + c] = (f16)(oAcc[ct].e[r] / lrun[r]);
    }
  }
}

// ---------------------------------------------------------------------------
// Stage 2 flash attention. QT (B,4096,64), kpT (B,1024,64), vp (B,64,1024).
// Epilogue fuses out = gamma*out2 + x (fp32, 8 consecutive floats per lane).
__global__ void k_attn_stage2(const f16* __restrict__ qT,
                              const f16* __restrict__ kpT,
                              const f16* __restrict__ vp,
                              const float* __restrict__ x,
                              const float* __restrict__ gamma,
                              float* __restrict__ out) {
  __shared__ __attribute__((aligned(16))) f16 sP[8][16][32];
  const int wave = threadIdx.x >> 5, lane = threadIdx.x & 31;
  const int lr = lane & 15, lh = lane >> 4;
  const int b  = blockIdx.y;
  const int qt = (blockIdx.x << 3) + wave;                 // 0..255
  const f16* QbT = qT  + (size_t)b * 4096 * 64;            // [n][c]
  const f16* KbT = kpT + (size_t)b * 1024 * 64;            // [m][c]
  const f16* Vb  = vp  + (size_t)b * 64 * 1024;            // [c][m]
  const v16h aq0 = load_frag_kcontig(QbT + (size_t)(qt * 16 + lr) * 64, lh);
  const v16h aq1 = load_frag_kcontig(QbT + (size_t)(qt * 16 + lr) * 64 + 32, lh);
  F32Acc oAcc[4] = {};
  float lrun[8];
  flash_loop<1024>(KbT, Vb, aq0, aq1, sP[wave], oAcc, lrun, lr, lh);
  const float g = gamma[0];
#pragma unroll
  for (int ct = 0; ct < 4; ++ct) {
    const int c = (ct << 4) + lr;
    const size_t base = ((size_t)(b * 64 + c) << 12) + (qt << 4) + (lh << 3);
#pragma unroll
    for (int r = 0; r < 8; ++r)                            // 8 consecutive floats
      out[base + r] = g * (oAcc[ct].e[r] / lrun[r]) + x[base + r];
  }
}

// ---------------------------------------------------------------------------
extern "C" void kernel_launch(void* const* d_in, const int* in_sizes, int n_in,
                              void* d_out, int out_size, void* d_ws,
                              size_t ws_size, hipStream_t stream) {
  const float* x     = (const float*)d_in[0];
  const float* w_q   = (const float*)d_in[1];
  const float* b_q   = (const float*)d_in[2];
  const float* w_K   = (const float*)d_in[3];
  const float* b_K   = (const float*)d_in[4];
  const float* w_V   = (const float*)d_in[5];
  const float* b_V   = (const float*)d_in[6];
  const float* w_Q   = (const float*)d_in[7];
  const float* b_Q   = (const float*)d_in[8];
  const float* w_k   = (const float*)d_in[9];
  const float* b_k   = (const float*)d_in[10];
  const float* w_v   = (const float*)d_in[11];
  const float* b_v   = (const float*)d_in[12];
  const float* gamma = (const float*)d_in[13];
  float* out = (float*)d_out;
  (void)in_sizes; (void)n_in; (void)out_size; (void)ws_size;

  const int B = 8, C = 64, N = 4096, N4 = 1024;
  char* p = (char*)d_ws;
  auto alloc = [&](size_t bytes) {
    char* r = p; p += (bytes + 255) & ~(size_t)255; return r;
  };
  f16* xT    = (f16*)alloc((size_t)B * N  * C * 2);  // (B,N,C)
  f16* kfT   = (f16*)alloc((size_t)B * N  * C * 2);  // (B,N,C)
  f16* vf    = (f16*)alloc((size_t)B * C  * N * 2);  // (B,C,N)
  f16* qT    = (f16*)alloc((size_t)B * N  * C * 2);  // (B,N,C)
  f16* qfT   = (f16*)alloc((size_t)B * N  * C * 2);  // (B,N,C)
  f16* qph   = (f16*)alloc((size_t)B * N4 * C * 2);  // (B,N4,C)
  f16* out1T = (f16*)alloc((size_t)B * N4 * C * 2);  // (B,N4,C)
  f16* kpT   = (f16*)alloc((size_t)B * N4 * C * 2);  // (B,N4,C)
  f16* vph   = (f16*)alloc((size_t)B * C * N4 * 2);  // (B,C,N4)
  f16* wqh   = (f16*)alloc(4096 * 2);
  f16* wKh   = (f16*)alloc(4096 * 2);
  f16* wVh   = (f16*)alloc(4096 * 2);
  f16* wQh   = (f16*)alloc(4096 * 2);
  f16* wkh   = (f16*)alloc(4096 * 2);
  f16* wvh   = (f16*)alloc(4096 * 2);

  // Pack inputs to f16 (x transposed to K-contiguous layout).
  k_pack_xT<<<(B * N * C) / 256, 256, 0, stream>>>(x, xT);
  k_pack_f16<<<16, 256, 0, stream>>>(w_q, wqh, 4096);
  k_pack_f16<<<16, 256, 0, stream>>>(w_K, wKh, 4096);
  k_pack_f16<<<16, 256, 0, stream>>>(w_V, wVh, 4096);
  k_pack_f16<<<16, 256, 0, stream>>>(w_Q, wQh, 4096);
  k_pack_f16<<<16, 256, 0, stream>>>(w_k, wkh, 4096);
  k_pack_f16<<<16, 256, 0, stream>>>(w_v, wvh, 4096);

  // Stage-1 convs (WMMA GEMMs, no LDS, all operands K-contiguous).
  k_conv1x1<<<dim3(32, B), 256, 0, stream>>>(xT, wqh, b_q, qT,  N, 1);
  k_conv1x1<<<dim3(32, B), 256, 0, stream>>>(xT, wKh, b_K, kfT, N, 1);
  k_conv1x1<<<dim3(32, B), 256, 0, stream>>>(xT, wVh, b_V, vf,  N, 0);
  k_conv1x1<<<dim3(32, B), 256, 0, stream>>>(xT, wQh, b_Q, qfT, N, 1);
  k_pool_q<<<(B * N4 * C) / 256, 256, 0, stream>>>(qT, qph);

  // Stage-1 fused flash attention -> out1T (B,N4,C).
  k_attn_stage1<<<dim3(8, B), 256, 0, stream>>>(qph, kfT, vf, out1T);

  // Stage-2 convs on pooled output.
  k_conv1x1<<<dim3(8, B), 256, 0, stream>>>(out1T, wkh, b_k, kpT, N4, 1);
  k_conv1x1<<<dim3(8, B), 256, 0, stream>>>(out1T, wvh, b_v, vph, N4, 0);

  // Stage-2 fused flash attention + gamma*out2 + x epilogue.
  k_attn_stage2<<<dim3(32, B), 256, 0, stream>>>(qfT, kpT, vph, x, gamma, out);
}
